// RecurrentEncoder_72516227826192
// MI455X (gfx1250) — compile-verified, matrix operand order checked by reading, not run on hardware
//
#include <hip/hip_runtime.h>
#include <hip/hip_bf16.h>
#include <math.h>

typedef _Float16 v16h __attribute__((ext_vector_type(16)));
typedef _Float16 v8h  __attribute__((ext_vector_type(8)));
typedef float    v8f  __attribute__((ext_vector_type(8)));

constexpr int       BB    = 8;
constexpr int       NNODE = 30;
constexpr int       TT    = 100;
constexpr int       FIN   = 4;
constexpr int       HH    = 128;
constexpr int       EE    = 870;           // N*(N-1)
constexpr int       OO    = 2;
constexpr int       SS    = BB * EE;       // 6960 sequences
constexpr long long RE    = (long long)SS * TT;        // 696000 edge rows
constexpr long long RN    = (long long)BB * NNODE * TT; // 24000 node rows

// ---------------------------------------------------------------------------
// WMMA GEMM:  out[M x Ntot] = act( A16[M x K] @ W16[Ntot x K]^T + bias )
// A and W are K-contiguous f16 (PyTorch (out,in) weight layout).
// Each wave computes 16 rows x 128 cols (8 accumulators); blockDim=256 (8 waves),
// grid.y picks a 128-wide N chunk.  M must be a multiple of 16 (true here).
// Fragment layout follows the CDNA5 ISA 16-bit A-matrix table:
//   lanes 0-15 : row = lane,     halves = K[0..7]  and K[16..23]
//   lanes 16-31: row = lane-16,  halves = K[8..15] and K[24..31]
// B mirrors A with columns (W row = output column, K-contiguous).
// C/D: lane&15 = column, (lane>>4)*8 + v = row.
// K is a template parameter so the K-loop fully unrolls: all fragment loads
// become base-pointer + immediate-offset (no per-iteration 64-bit address
// adds -> fewer WMMA->VALU hazard NOPs, loads hoistable across WMMAs).
// ---------------------------------------------------------------------------
template <int K>
__global__ void wmma_gemm_k(const _Float16* __restrict__ A,
                            const _Float16* __restrict__ W,
                            const float* __restrict__ bias,
                            float* __restrict__ outF,
                            _Float16* __restrict__ out16,
                            int M, int Ntot, int act) {
  const int wave = threadIdx.x >> 5;
  const int lane = threadIdx.x & 31;
  const int half = lane >> 4;
  const int r    = lane & 15;
  const int row0 = (blockIdx.x * 8 + wave) * 16;
  if (row0 >= M) return;                      // wave-uniform: EXEC all-ones inside
  const int nBase = blockIdx.y * 128;

  const _Float16* abase = A + (size_t)(row0 + r) * K + half * 8;
  const _Float16* wbase[8];
#pragma unroll
  for (int j = 0; j < 8; ++j)
    wbase[j] = W + (size_t)(nBase + j * 16 + r) * K + half * 8;

  v8f acc[8];
#pragma unroll
  for (int j = 0; j < 8; ++j)
#pragma unroll
    for (int v = 0; v < 8; ++v) acc[j][v] = 0.0f;

#pragma unroll
  for (int kk = 0; kk < K; kk += 32) {
    v8h alo = *(const v8h*)(abase + kk);
    v8h ahi = *(const v8h*)(abase + kk + 16);
    v16h af = __builtin_shufflevector(alo, ahi, 0,1,2,3,4,5,6,7,8,9,10,11,12,13,14,15);

    v16h wf[8];
#pragma unroll
    for (int j = 0; j < 8; ++j) {
      v8h wlo = *(const v8h*)(wbase[j] + kk);
      v8h whi = *(const v8h*)(wbase[j] + kk + 16);
      wf[j] = __builtin_shufflevector(wlo, whi, 0,1,2,3,4,5,6,7,8,9,10,11,12,13,14,15);
    }
#pragma unroll
    for (int j = 0; j < 8; ++j)
      acc[j] = __builtin_amdgcn_wmma_f32_16x16x32_f16(
          false, af, false, wf[j], (short)0, acc[j], false, false);
  }

#pragma unroll
  for (int j = 0; j < 8; ++j) {
    const int col = nBase + j * 16 + r;
    const float bv = bias ? bias[col] : 0.0f;
#pragma unroll
    for (int v = 0; v < 8; ++v) {
      const int m = row0 + half * 8 + v;
      float x = acc[j][v] + bv;
      if (act) x = (x > 0.0f) ? x : (expf(x) - 1.0f);   // ELU
      const size_t idx = (size_t)m * Ntot + col;
      if (outF)  outF[idx]  = x;
      if (out16) out16[idx] = (_Float16)x;
    }
  }
}

// ---------------------------------------------------------------------------
// Fused GRU step: gh = h16 @ whh^T + bhh via WMMA into LDS, then gates.
// Block = 256 threads (8 waves) owns 16 sequences (rows). Wave w computes
// column tiles [w*48, w*48+48) of the 384-wide gh into LDS; after the barrier
// the whole block applies the GRU gate math for its 16x128 state slice.
// Rows are block-disjoint, and all h16 reads (WMMA phase) happen before the
// barrier, so the in-kernel h/h16 update is safe.
// ---------------------------------------------------------------------------
__global__ void gru_fused_k(const _Float16* __restrict__ gi,
                            const _Float16* __restrict__ whh,
                            const float* __restrict__ bhh,
                            float* __restrict__ h, _Float16* __restrict__ h16,
                            float* __restrict__ ys, int t) {
  __shared__ float gh[16 * 384];
  const int wave = threadIdx.x >> 5;
  const int lane = threadIdx.x & 31;
  const int half = lane >> 4;
  const int r    = lane & 15;
  const int r0   = blockIdx.x * 16;
  const int nb   = wave * 48;

  const _Float16* abase = h16 + (size_t)(r0 + r) * HH + half * 8;
  const _Float16* wbase[3];
#pragma unroll
  for (int j = 0; j < 3; ++j)
    wbase[j] = whh + (size_t)(nb + j * 16 + r) * HH + half * 8;

  v8f acc[3];
#pragma unroll
  for (int j = 0; j < 3; ++j)
#pragma unroll
    for (int v = 0; v < 8; ++v) acc[j][v] = 0.0f;

#pragma unroll
  for (int kk = 0; kk < HH; kk += 32) {
    v8h alo = *(const v8h*)(abase + kk);
    v8h ahi = *(const v8h*)(abase + kk + 16);
    v16h af = __builtin_shufflevector(alo, ahi, 0,1,2,3,4,5,6,7,8,9,10,11,12,13,14,15);
    v16h wf[3];
#pragma unroll
    for (int j = 0; j < 3; ++j) {
      v8h wlo = *(const v8h*)(wbase[j] + kk);
      v8h whi = *(const v8h*)(wbase[j] + kk + 16);
      wf[j] = __builtin_shufflevector(wlo, whi, 0,1,2,3,4,5,6,7,8,9,10,11,12,13,14,15);
    }
#pragma unroll
    for (int j = 0; j < 3; ++j)
      acc[j] = __builtin_amdgcn_wmma_f32_16x16x32_f16(
          false, af, false, wf[j], (short)0, acc[j], false, false);
  }

#pragma unroll
  for (int j = 0; j < 3; ++j) {
    const int col = nb + j * 16 + r;
    const float bv = bhh[col];
#pragma unroll
    for (int v = 0; v < 8; ++v)
      gh[(half * 8 + v) * 384 + col] = acc[j][v] + bv;
  }
  __syncthreads();

  // gates: 16 rows x 128 state cols = 2048 elements, 8 per thread
#pragma unroll
  for (int i = 0; i < 8; ++i) {
    const int idx = threadIdx.x + i * 256;
    const int row = idx >> 7;
    const int c   = idx & (HH - 1);
    const int s   = r0 + row;
    const long long gib = ((long long)s * TT + t) * (3 * HH);
    const float ir = (float)gi[gib + c];
    const float iz = (float)gi[gib + HH + c];
    const float in = (float)gi[gib + 2 * HH + c];
    const float hr_ = gh[row * 384 + c];
    const float hz  = gh[row * 384 + HH + c];
    const float hn  = gh[row * 384 + 2 * HH + c];
    const float rg = 1.0f / (1.0f + expf(-(ir + hr_)));
    const float zg = 1.0f / (1.0f + expf(-(iz + hz)));
    const float ng = tanhf(in + rg * hn);
    const long long hg = (long long)s * HH + c;
    const float hv = (1.0f - zg) * ng + zg * h[hg];
    h[hg]  = hv;
    h16[hg] = (_Float16)hv;
    ys[((long long)s * TT + t) * HH + c] = hv;
  }
}

// ---------------------------------------------------------------------------
// Small helper kernels
// ---------------------------------------------------------------------------
__global__ void conv_f16_k(const float* __restrict__ s, _Float16* __restrict__ d, long long n) {
  long long i = (long long)blockIdx.x * blockDim.x + threadIdx.x;
  if (i < n) d[i] = (_Float16)s[i];
}
__global__ void zero_f32_k(float* p, long long n) {
  long long i = (long long)blockIdx.x * blockDim.x + threadIdx.x;
  if (i < n) p[i] = 0.0f;
}
__global__ void zero_f16_k(_Float16* p, long long n) {
  long long i = (long long)blockIdx.x * blockDim.x + threadIdx.x;
  if (i < n) p[i] = (_Float16)0.0f;
}
__global__ void copy_f32_k(const float* __restrict__ s, float* __restrict__ d, long long n) {
  long long i = (long long)blockIdx.x * blockDim.x + threadIdx.x;
  if (i < n) d[i] = s[i];
}

// Decode one-hot rel_send / rel_rec into index tables.
__global__ void build_idx_k(const float* __restrict__ rel_rec,
                            const float* __restrict__ rel_send,
                            int* __restrict__ send_idx, int* __restrict__ recv_idx,
                            int* __restrict__ recv_list) {
  int i = blockIdx.x * blockDim.x + threadIdx.x;
  if (i < EE) {
    int s = 0, r = 0;
    for (int n = 0; n < NNODE; ++n) {
      if (rel_send[i * NNODE + n] > 0.5f) s = n;
      if (rel_rec [i * NNODE + n] > 0.5f) r = n;
    }
    send_idx[i] = s; recv_idx[i] = r;
  }
  if (i < NNODE) {
    int cnt = 0;
    for (int e = 0; e < EE; ++e)
      if (rel_rec[e * NNODE + i] > 0.5f) recv_list[i * (NNODE - 1) + cnt++] = e;
  }
}

// MLP1 layer1: (RN x 4) @ (128 x 4)^T + b, ELU -> f16.  K=4 too small for WMMA.
__global__ void mlp1_layer1_k(const float* __restrict__ x, const float* __restrict__ w,
                              const float* __restrict__ b, _Float16* __restrict__ out) {
  long long gid = (long long)blockIdx.x * blockDim.x + threadIdx.x;
  if (gid >= RN * HH) return;
  int c = (int)(gid % HH);
  long long row = gid / HH;
  const float* xr = x + row * FIN;
  const float* wr = w + (long long)c * FIN;
  float s = b[c];
#pragma unroll
  for (int k = 0; k < FIN; ++k) s += xr[k] * wr[k];
  s = (s > 0.0f) ? s : (expf(s) - 1.0f);
  out[gid] = (_Float16)s;
}

// Per-feature sum / sum-of-squares over M rows of h[M x 128].
__global__ void reduce_stats_k(const float* __restrict__ h, int M, float* __restrict__ stats) {
  int c = threadIdx.x & 127;
  int rstart  = blockIdx.x * (blockDim.x >> 7) + (threadIdx.x >> 7);
  int rstride = gridDim.x * (blockDim.x >> 7);
  float s = 0.0f, s2 = 0.0f;
  for (int m = rstart; m < M; m += rstride) {
    float v = h[(size_t)m * HH + c];
    s += v; s2 += v * v;
  }
  atomicAdd(&stats[c], s);
  atomicAdd(&stats[HH + c], s2);
}

__global__ void normalize_k(const float* __restrict__ h, const float* __restrict__ stats,
                            const float* __restrict__ g, const float* __restrict__ bt,
                            float* __restrict__ outF, _Float16* __restrict__ out16, int M) {
  long long gid = (long long)blockIdx.x * blockDim.x + threadIdx.x;
  if (gid >= (long long)M * HH) return;
  int c = (int)(gid & (HH - 1));
  float mu  = stats[c] / (float)M;
  float var = stats[HH + c] / (float)M - mu * mu;
  float x = (h[gid] - mu) * rsqrtf(var + 1e-5f) * g[c] + bt[c];
  if (outF)  outF[gid]  = x;
  if (out16) out16[gid] = (_Float16)x;
}

// node2edge: out[b,e,t, 0:128]=x[send(e)], [128:256]=x[recv(e)]; 8 halves/thread.
__global__ void node2edge_k(const _Float16* __restrict__ xn,
                            const int* __restrict__ send_idx, const int* __restrict__ recv_idx,
                            _Float16* __restrict__ out) {
  long long gid = (long long)blockIdx.x * blockDim.x + threadIdx.x;
  if (gid >= RE * 32) return;
  long long er = gid >> 5;
  int c0 = (int)(gid & 31) * 8;
  int t = (int)(er % TT);
  long long be = er / TT;
  int e = (int)(be % EE);
  int b = (int)(be / EE);
  int n  = (c0 < HH) ? send_idx[e] : recv_idx[e];
  int h0 = c0 & (HH - 1);
  const v8h* src = (const v8h*)(xn + (((long long)(b * NNODE + n) * TT + t) * HH + h0));
  *(v8h*)(out + er * (2 * HH) + c0) = *src;
}

// concat for MLP4 input: [0:256] = node2edge(x3), [256:384] = x_skip (f16).
__global__ void concat_e4_k(const _Float16* __restrict__ x3,
                            const _Float16* __restrict__ xskip16,
                            const int* __restrict__ send_idx, const int* __restrict__ recv_idx,
                            _Float16* __restrict__ out) {
  long long gid = (long long)blockIdx.x * blockDim.x + threadIdx.x;
  if (gid >= RE * 48) return;
  long long er = gid / 48;
  int c0 = (int)(gid % 48) * 8;
  const v8h* src;
  if (c0 < 2 * HH) {
    int t = (int)(er % TT);
    long long be = er / TT;
    int e = (int)(be % EE);
    int b = (int)(be / EE);
    int n  = (c0 < HH) ? send_idx[e] : recv_idx[e];
    int h0 = c0 & (HH - 1);
    src = (const v8h*)(x3 + (((long long)(b * NNODE + n) * TT + t) * HH + h0));
  } else {
    src = (const v8h*)(xskip16 + er * HH + (c0 - 2 * HH));
  }
  *(v8h*)(out + er * (3 * HH) + c0) = *src;
}

// edge2node: x_node[b,n,t,:] = (1/N) * sum over the 29 incoming edges of node n.
__global__ void edge2node_k(const float* __restrict__ xe, const int* __restrict__ recv_list,
                            _Float16* __restrict__ out) {
  long long gid = (long long)blockIdx.x * blockDim.x + threadIdx.x;
  if (gid >= RN * HH) return;
  int hcol = (int)(gid % HH);
  long long rn = gid / HH;
  int t = (int)(rn % TT);
  long long bn = rn / TT;
  int n = (int)(bn % NNODE);
  int b = (int)(bn / NNODE);
  float s = 0.0f;
  for (int j = 0; j < NNODE - 1; ++j) {
    int e = recv_list[n * (NNODE - 1) + j];
    s += xe[(((long long)(b * EE + e)) * TT + t) * HH + hcol];
  }
  out[gid] = (_Float16)(s / (float)NNODE);
}

__global__ void proj_prior_k(const float* __restrict__ ys, const float* __restrict__ pw,
                             const float* __restrict__ pb, float* __restrict__ out) {
  long long gid = (long long)blockIdx.x * blockDim.x + threadIdx.x;
  if (gid >= (long long)SS * TT * OO) return;
  int o = (int)(gid & 1);
  long long st = gid >> 1;
  const float* yr = ys + st * HH;
  const float* wr = pw + (long long)o * HH;
  float acc = pb[o];
  for (int k = 0; k < HH; ++k) acc += yr[k] * wr[k];
  out[gid] = acc;
}

__global__ void proj_enc_k(const float* __restrict__ ysF, const float* __restrict__ ysR,
                           const float* __restrict__ ew, const float* __restrict__ eb,
                           float* __restrict__ out) {
  long long gid = (long long)blockIdx.x * blockDim.x + threadIdx.x;
  if (gid >= (long long)SS * TT * OO) return;
  int o = (int)(gid & 1);
  long long st = gid >> 1;
  const float* w = ew + (long long)o * (2 * HH);
  float acc = eb[o];
  for (int k = 0; k < HH; ++k)
    acc += ysF[st * HH + k] * w[k] + ysR[st * HH + k] * w[HH + k];
  out[gid] = acc;
}

// ---------------------------------------------------------------------------
extern "C" void kernel_launch(void* const* d_in, const int* in_sizes, int n_in,
                              void* d_out, int out_size, void* d_ws, size_t ws_size,
                              hipStream_t stream) {
  (void)in_sizes; (void)n_in; (void)out_size; (void)ws_size;

  const float* inputs   = (const float*)d_in[0];
  const float* rel_rec  = (const float*)d_in[1];
  const float* rel_send = (const float*)d_in[2];
  const float *mw1[4], *mb1[4], *mw2[4], *mb2[4], *mg[4], *mbt[4];
  for (int k = 0; k < 4; ++k) {
    int b = 3 + 6 * k;
    mw1[k] = (const float*)d_in[b];     mb1[k] = (const float*)d_in[b + 1];
    mw2[k] = (const float*)d_in[b + 2]; mb2[k] = (const float*)d_in[b + 3];
    mg[k]  = (const float*)d_in[b + 4]; mbt[k] = (const float*)d_in[b + 5];
  }
  const float* gf_wih = (const float*)d_in[27];
  const float* gf_whh = (const float*)d_in[28];
  const float* gf_bih = (const float*)d_in[29];
  const float* gf_bhh = (const float*)d_in[30];
  const float* gr_wih = (const float*)d_in[31];
  const float* gr_whh = (const float*)d_in[32];
  const float* gr_bih = (const float*)d_in[33];
  const float* gr_bhh = (const float*)d_in[34];
  const float* pw = (const float*)d_in[35];
  const float* pb = (const float*)d_in[36];
  const float* ew = (const float*)d_in[37];
  const float* eb = (const float*)d_in[38];

  // --- workspace bump allocator (256B aligned) ---
  char* base = (char*)d_ws;
  size_t off = 0;
  auto take = [&](size_t bytes) -> void* {
    off = (off + 255) & ~(size_t)255;
    void* r = base + off;
    off += bytes;
    return r;
  };

  int* send_idx  = (int*)take((size_t)EE * 4);
  int* recv_idx  = (int*)take((size_t)EE * 4);
  int* recv_list = (int*)take((size_t)NNODE * (NNODE - 1) * 4);
  float* stats   = (float*)take(256 * 4);

  // f16 weights
  _Float16* w16_m1w2 = (_Float16*)take((size_t)HH * HH * 2);
  _Float16* w16_m2w1 = (_Float16*)take((size_t)HH * 2 * HH * 2);
  _Float16* w16_m2w2 = (_Float16*)take((size_t)HH * HH * 2);
  _Float16* w16_m3w1 = (_Float16*)take((size_t)HH * HH * 2);
  _Float16* w16_m3w2 = (_Float16*)take((size_t)HH * HH * 2);
  _Float16* w16_m4w1 = (_Float16*)take((size_t)HH * 3 * HH * 2);
  _Float16* w16_m4w2 = (_Float16*)take((size_t)HH * HH * 2);
  _Float16* w16_gfwih = (_Float16*)take((size_t)3 * HH * HH * 2);
  _Float16* w16_gfwhh = (_Float16*)take((size_t)3 * HH * HH * 2);
  _Float16* w16_grwih = (_Float16*)take((size_t)3 * HH * HH * 2);
  _Float16* w16_grwhh = (_Float16*)take((size_t)3 * HH * HH * 2);

  // node-scale buffers
  _Float16* n_h16 = (_Float16*)take((size_t)RN * HH * 2);
  float*    n_f32 = (float*)take((size_t)RN * HH * 4);
  _Float16* x1_16 = (_Float16*)take((size_t)RN * HH * 2);
  _Float16* n3_16 = (_Float16*)take((size_t)RN * HH * 2);
  _Float16* x3_16 = (_Float16*)take((size_t)RN * HH * 2);

  // GRU state
  float*    hf   = (float*)take((size_t)SS * HH * 4);
  _Float16* hf16 = (_Float16*)take((size_t)SS * HH * 2);
  float*    hr   = (float*)take((size_t)SS * HH * 4);
  _Float16* hr16 = (_Float16*)take((size_t)SS * HH * 2);

  // big regions (stage-reused)
  char* Abuf = (char*)take((size_t)RE * 3 * HH * 2); // MLP2 hid f32 | e_in4 f16 | gi_f f16
  char* Bbuf = (char*)take((size_t)RE * 3 * HH * 2); // x2 f32 | MLP4 hid f32 | gi_r f16
  char* Cbuf = (char*)take((size_t)RE * 2 * HH * 2); // e2 f16 | ys_fwd f32
  char* Dbuf = (char*)take((size_t)RE * HH * 4);     // MLP h2 f32 | ys_rev f32
  _Float16* hid16 = (_Float16*)take((size_t)RE * HH * 2); // MLP2/MLP4 hidden f16
  _Float16* x2_16 = (_Float16*)take((size_t)RE * HH * 2); // skip connection f16
  _Float16* x4_16 = (_Float16*)take((size_t)RE * HH * 2); // GRU input f16

  auto cdiv = [](long long a, long long b) { return (unsigned)((a + b - 1) / b); };
  auto conv = [&](const float* s, _Float16* d, long long n) {
    conv_f16_k<<<cdiv(n, 256), 256, 0, stream>>>(s, d, n);
  };
  auto gemm = [&](const _Float16* A16, const _Float16* W16, const float* bias,
                  float* oF, _Float16* o16, long long M, int Ntot, int K, int act) {
    dim3 g(cdiv(M, 128), Ntot / 128);
    if (K == 128)
      wmma_gemm_k<128><<<g, 256, 0, stream>>>(A16, W16, bias, oF, o16, (int)M, Ntot, act);
    else if (K == 256)
      wmma_gemm_k<256><<<g, 256, 0, stream>>>(A16, W16, bias, oF, o16, (int)M, Ntot, act);
    else
      wmma_gemm_k<384><<<g, 256, 0, stream>>>(A16, W16, bias, oF, o16, (int)M, Ntot, act);
  };
  auto statsnorm = [&](const float* h2, long long M, const float* g, const float* bt,
                       float* oF, _Float16* o16) {
    zero_f32_k<<<1, 256, 0, stream>>>(stats, 256);
    reduce_stats_k<<<512, 256, 0, stream>>>(h2, (int)M, stats);
    normalize_k<<<cdiv(M * HH, 256), 256, 0, stream>>>(h2, stats, g, bt, oF, o16, (int)M);
  };

  // --- index tables + weight conversion ---
  build_idx_k<<<cdiv(EE, 256), 256, 0, stream>>>(rel_rec, rel_send, send_idx, recv_idx, recv_list);
  conv(mw2[0], w16_m1w2, (long long)HH * HH);
  conv(mw1[1], w16_m2w1, (long long)HH * 2 * HH);
  conv(mw2[1], w16_m2w2, (long long)HH * HH);
  conv(mw1[2], w16_m3w1, (long long)HH * HH);
  conv(mw2[2], w16_m3w2, (long long)HH * HH);
  conv(mw1[3], w16_m4w1, (long long)HH * 3 * HH);
  conv(mw2[3], w16_m4w2, (long long)HH * HH);
  conv(gf_wih, w16_gfwih, (long long)3 * HH * HH);
  conv(gf_whh, w16_gfwhh, (long long)3 * HH * HH);
  conv(gr_wih, w16_grwih, (long long)3 * HH * HH);
  conv(gr_whh, w16_grwhh, (long long)3 * HH * HH);

  // --- MLP1 (node scale) ---
  mlp1_layer1_k<<<cdiv(RN * HH, 256), 256, 0, stream>>>(inputs, mw1[0], mb1[0], n_h16);
  gemm(n_h16, w16_m1w2, mb2[0], n_f32, nullptr, RN, HH, HH, 1);
  statsnorm(n_f32, RN, mg[0], mbt[0], nullptr, x1_16);

  // --- node2edge -> MLP2 (edge scale) ---
  _Float16* e2_16 = (_Float16*)Cbuf;
  node2edge_k<<<cdiv(RE * 32, 256), 256, 0, stream>>>(x1_16, send_idx, recv_idx, e2_16);
  gemm(e2_16, w16_m2w1, mb1[1], (float*)Abuf, hid16, RE, HH, 2 * HH, 1);
  gemm(hid16, w16_m2w2, mb2[1], (float*)Dbuf, nullptr, RE, HH, HH, 1);
  statsnorm((float*)Dbuf, RE, mg[1], mbt[1], (float*)Bbuf /*x2 f32*/, x2_16);

  // --- edge2node -> MLP3 (node scale) ---
  edge2node_k<<<cdiv(RN * HH, 256), 256, 0, stream>>>((const float*)Bbuf, recv_list, n3_16);
  gemm(n3_16, w16_m3w1, mb1[2], n_f32, n_h16, RN, HH, HH, 1);
  gemm(n_h16, w16_m3w2, mb2[2], n_f32, nullptr, RN, HH, HH, 1);
  statsnorm(n_f32, RN, mg[2], mbt[2], nullptr, x3_16);

  // --- concat [node2edge(x3), x_skip] -> MLP4 ---
  _Float16* e4_16 = (_Float16*)Abuf;
  concat_e4_k<<<cdiv(RE * 48, 256), 256, 0, stream>>>(x3_16, x2_16, send_idx, recv_idx, e4_16);
  gemm(e4_16, w16_m4w1, mb1[3], (float*)Bbuf, hid16, RE, HH, 3 * HH, 1);
  gemm(hid16, w16_m4w2, mb2[3], (float*)Dbuf, nullptr, RE, HH, HH, 1);
  statsnorm((float*)Dbuf, RE, mg[3], mbt[3], nullptr, x4_16);

  // --- GRU input precompute: gi = x4 @ w_ih^T + b_ih (f16 store) ---
  _Float16* gi_f = (_Float16*)Abuf;  // e_in4 dead after MLP4 g1
  _Float16* gi_r = (_Float16*)Bbuf;  // MLP4 hidden f32 dead
  gemm(x4_16, w16_gfwih, gf_bih, nullptr, gi_f, RE, 3 * HH, HH, 0);
  gemm(x4_16, w16_grwih, gr_bih, nullptr, gi_r, RE, 3 * HH, HH, 0);

  float* ysF = (float*)Cbuf;  // e2 dead
  float* ysR = (float*)Dbuf;  // MLP4 h2 dead after norm

  // --- forward GRU (fused WMMA + gates, one kernel per step) ---
  zero_f32_k<<<cdiv((long long)SS * HH, 256), 256, 0, stream>>>(hf, (long long)SS * HH);
  zero_f16_k<<<cdiv((long long)SS * HH, 256), 256, 0, stream>>>(hf16, (long long)SS * HH);
  for (int t = 0; t < TT; ++t)
    gru_fused_k<<<SS / 16, 256, 0, stream>>>(gi_f, w16_gfwhh, gf_bhh, hf, hf16, ysF, t);

  // --- reverse GRU (scan t = T-1..0; equals reverse-in / reverse-out) ---
  zero_f32_k<<<cdiv((long long)SS * HH, 256), 256, 0, stream>>>(hr, (long long)SS * HH);
  zero_f16_k<<<cdiv((long long)SS * HH, 256), 256, 0, stream>>>(hr16, (long long)SS * HH);
  for (int st = 0; st < TT; ++st) {
    int t = TT - 1 - st;
    gru_fused_k<<<SS / 16, 256, 0, stream>>>(gi_r, w16_grwhh, gr_bhh, hr, hr16, ysR, t);
  }

  // --- outputs: prior | enc | hT ---
  float* out = (float*)d_out;
  const long long prior_n = (long long)SS * TT * OO;
  proj_prior_k<<<cdiv(prior_n, 256), 256, 0, stream>>>(ysF, pw, pb, out);
  proj_enc_k<<<cdiv(prior_n, 256), 256, 0, stream>>>(ysF, ysR, ew, eb, out + prior_n);
  copy_f32_k<<<cdiv((long long)SS * HH, 256), 256, 0, stream>>>(hf, out + 2 * prior_n,
                                                                (long long)SS * HH);
}
